// PointNetPPBoxEstimationCuda_80599356277052
// MI455X (gfx1250) — matrix operand order repaired and checked by reading, not compiled, not run on hardware
//
#include <hip/hip_runtime.h>
#include <stdint.h>

// ---------------------------------------------------------------------------
// PointNet++ box-estimation head for MI455X (gfx1250, wave32).
//
// Design notes (compile-only environment; reasoning from HW specs):
//  * All einsums/FCs are GEMMs vs. tiny weights -> bf16 WMMA (16x16x32, f32 acc)
//    is the fast path; total matmul FLOPs (~126G) are negligible vs WMMA peak,
//    so the pipeline is bound by ~1.8GB of activation traffic (~80us @23.3TB/s).
//    Keeping activations bf16 halves that traffic.
//  * BatchNorm stats (mean/var over ALL rows) are computed in the GEMM epilogue
//    as per-block partials reduced deterministically by a finalize kernel
//    (no float atomics -> bitwise deterministic across graph replays).
//  * Weight panels (16 output rows x full K) are staged to LDS with the Tensor
//    Data Mover (tensor_load_to_lds + s_wait_tensorcnt), descriptor built per
//    CDNA5 ISA. This toolchain exposes the 6-arg builtin.
//  * Workspace: ~830MB (two bf16 ping-pong activation buffers dominate).
// ---------------------------------------------------------------------------

#define BB 256
#define NPTS 2048
#define MAXK 544   // fc1 K = 514 padded to 544

typedef __bf16 bf16;
typedef __bf16 v16bf  __attribute__((ext_vector_type(16)));
typedef __bf16 bf16x8 __attribute__((ext_vector_type(8)));
typedef float  v8f    __attribute__((ext_vector_type(8)));
typedef unsigned int u32x4 __attribute__((ext_vector_type(4)));
typedef int    i32x8  __attribute__((ext_vector_type(8)));
typedef int    i32x4  __attribute__((ext_vector_type(4)));

// -------------------------------- transpose (B,3,N) -> (B,N,3) --------------
__global__ void k_transpose(const float* __restrict__ pts, float* __restrict__ xyz) {
  long long t = (long long)blockIdx.x * 256 + threadIdx.x;
  const long long total = (long long)BB * NPTS * 3;
  if (t >= total) return;
  int d = (int)(t % 3);
  long long r = t / 3;
  int b = (int)(r / NPTS), n = (int)(r % NPTS);
  xyz[t] = pts[((long long)b * 3 + d) * NPTS + n];
}

// -------------------------------- farthest point sampling -------------------
// One block per batch. Matches reference scan: emit far, update dist, argmax
// (ties -> lowest index, like jnp.argmax).
__global__ __launch_bounds__(256) void k_fps(const float* __restrict__ xyz,
                                             int N, int S, int* __restrict__ out) {
  int b = blockIdx.x, tid = threadIdx.x;
  __shared__ float sx[2048], sy[2048], sz[2048], dist[2048];
  __shared__ float rbest[256];
  __shared__ int   rarg[256];
  __shared__ int   s_far;
  for (int i = tid; i < N; i += 256) {
    const float* p = xyz + ((long long)b * N + i) * 3;
    sx[i] = p[0]; sy[i] = p[1]; sz[i] = p[2];
    dist[i] = 1e10f;
  }
  if (tid == 0) s_far = 0;
  __syncthreads();
  for (int it = 0; it < S; ++it) {
    int far = s_far;
    if (tid == 0) out[b * S + it] = far;
    float cx = sx[far], cy = sy[far], cz = sz[far];
    float best = -1.0f; int bi = 0x7FFFFFFF;
    for (int i = tid; i < N; i += 256) {
      float dx = sx[i] - cx, dy = sy[i] - cy, dz = sz[i] - cz;
      float d = dx * dx + dy * dy + dz * dz;
      float dm = fminf(dist[i], d);
      dist[i] = dm;
      if (dm > best) { best = dm; bi = i; }
    }
    rbest[tid] = best; rarg[tid] = bi;
    __syncthreads();
    for (int s = 128; s > 0; s >>= 1) {
      if (tid < s) {
        if (rbest[tid + s] > rbest[tid] ||
            (rbest[tid + s] == rbest[tid] && rarg[tid + s] < rarg[tid])) {
          rbest[tid] = rbest[tid + s]; rarg[tid] = rarg[tid + s];
        }
      }
      __syncthreads();
    }
    if (tid == 0) s_far = rarg[0];
    __syncthreads();
  }
}

// gather 3-vectors by index: dst[(b*S+s)*3+d] = src[(b*N+idx[b*S+s])*3+d]
__global__ void k_gather3(const float* __restrict__ src, const int* __restrict__ idx,
                          int N, int S, float* __restrict__ dst) {
  int t = blockIdx.x * 256 + threadIdx.x;
  int total = BB * S * 3;
  if (t >= total) return;
  int d = t % 3, r = t / 3, b = r / S;
  dst[t] = src[((long long)b * N + idx[r]) * 3 + d];
}

// -------------------------------- ball query --------------------------------
// First `ns` in-order indices with d2<r2, padded with the first hit
// (matches sort(where(mask, arange, n))[:ns] then replace-n-with-first).
__global__ void k_ballquery(const float* __restrict__ xyz, const float* __restrict__ cen,
                            int N, int S, float r2, int ns, int* __restrict__ idx) {
  int t = blockIdx.x * 256 + threadIdx.x;
  if (t >= BB * S) return;
  int b = t / S;
  float cx = cen[t * 3], cy = cen[t * 3 + 1], cz = cen[t * 3 + 2];
  const float* base = xyz + (long long)b * N * 3;
  int* o = idx + (long long)t * ns;
  int cnt = 0, first = 0;
  for (int n = 0; n < N && cnt < ns; ++n) {
    float dx = base[n * 3] - cx, dy = base[n * 3 + 1] - cy, dz = base[n * 3 + 2] - cz;
    if (dx * dx + dy * dy + dz * dz < r2) {
      if (cnt == 0) first = n;
      o[cnt++] = n;
    }
  }
  for (; cnt < ns; ++cnt) o[cnt] = first;
}

// -------------------------------- grouping kernels (build bf16 GEMM inputs) -
__global__ void k_group_sa1(const float* __restrict__ xyz, const float* __restrict__ l1x,
                            const int* __restrict__ idx, bf16* __restrict__ X) {
  int r = blockIdx.x * 256 + threadIdx.x;
  const int total = BB * 128 * 64;
  if (r >= total) return;
  int b = r / (128 * 64), s = (r / 64) % 128;
  int gi = idx[r];
  const float* p = xyz + ((long long)b * NPTS + gi) * 3;
  const float* c = l1x + ((long long)b * 128 + s) * 3;
  bf16* xr = X + (long long)r * 32;
  xr[0] = (bf16)(p[0] - c[0]); xr[1] = (bf16)(p[1] - c[1]); xr[2] = (bf16)(p[2] - c[2]);
  for (int k = 3; k < 32; ++k) xr[k] = (bf16)0.0f;
}

__global__ void k_group_sa2(const float* __restrict__ l1x, const float* __restrict__ l2x,
                            const int* __restrict__ idx, const float* __restrict__ l1f,
                            bf16* __restrict__ X) {
  int r = blockIdx.x * 256 + threadIdx.x;
  const int total = BB * 32 * 64;
  if (r >= total) return;
  int b = r / (32 * 64), s = (r / 64) % 32;
  int gi = idx[r];
  const float* p = l1x + ((long long)b * 128 + gi) * 3;
  const float* c = l2x + ((long long)b * 32 + s) * 3;
  bf16* xr = X + (long long)r * 160;
  xr[0] = (bf16)(p[0] - c[0]); xr[1] = (bf16)(p[1] - c[1]); xr[2] = (bf16)(p[2] - c[2]);
  const float* f = l1f + ((long long)b * 128 + gi) * 128;
  for (int k = 0; k < 128; ++k) xr[3 + k] = (bf16)f[k];
  for (int k = 131; k < 160; ++k) xr[k] = (bf16)0.0f;
}

__global__ void k_group_sa3(const float* __restrict__ l2x, const float* __restrict__ l2f,
                            bf16* __restrict__ X) {
  int r = blockIdx.x * 256 + threadIdx.x;
  if (r >= BB * 32) return;
  bf16* xr = X + (long long)r * 288;
  const float* p = l2x + (long long)r * 3;
  xr[0] = (bf16)p[0]; xr[1] = (bf16)p[1]; xr[2] = (bf16)p[2];
  const float* f = l2f + (long long)r * 256;
  for (int k = 0; k < 256; ++k) xr[3 + k] = (bf16)f[k];
  for (int k = 259; k < 288; ++k) xr[k] = (bf16)0.0f;
}

__global__ void k_fcin(const float* __restrict__ gf, const float* __restrict__ oh,
                       bf16* __restrict__ X) {
  int b = blockIdx.x * 64 + threadIdx.x;
  if (b >= BB) return;
  bf16* xr = X + (long long)b * 544;
  for (int c = 0; c < 512; ++c) xr[c] = (bf16)gf[b * 512 + c];
  xr[512] = (bf16)oh[b * 2]; xr[513] = (bf16)oh[b * 2 + 1];
  for (int k = 514; k < 544; ++k) xr[k] = (bf16)0.0f;
}

// -------------------------------- weight / bias packing (f32 -> padded bf16) -
__global__ void k_wpack(const float* __restrict__ src, bf16* __restrict__ dst,
                        int Cout, int Cin, int Cpad, int Kpad) {
  int t = blockIdx.x * 256 + threadIdx.x;
  if (t >= Cpad * Kpad) return;
  int o = t / Kpad, k = t % Kpad;
  float v = (o < Cout && k < Cin) ? src[(long long)o * Cin + k] : 0.0f;
  dst[t] = (bf16)v;
}

__global__ void k_bpack(const float* __restrict__ src, float* __restrict__ dst, int C, int Cpad) {
  int t = blockIdx.x * 256 + threadIdx.x;
  if (t >= Cpad) return;
  dst[t] = (t < C) ? src[t] : 0.0f;
}

// -------------------------------- WMMA GEMM ---------------------------------
// Y[M,C] = X[M,K] * Wt[C,K]^T (+bias). 8 waves/block, one 16x16 tile per wave
// stacked along M (128 rows per block). The 16-row weight panel (full K) is
// staged into LDS by the Tensor Data Mover (wave 0). Per-block BN partial
// sums/sumsq are emitted for deterministic two-stage reduction.
__global__ __launch_bounds__(256) void k_gemm(
    const bf16* __restrict__ X, const bf16* __restrict__ Wt,
    const float* __restrict__ bias,
    bf16* __restrict__ Y, float* __restrict__ Yf32,
    float* __restrict__ psum, float* __restrict__ psq,
    int nblkM, int M, int K, int C) {
  const int wave = threadIdx.x >> 5;
  const int lane = threadIdx.x & 31;
  const int hi   = lane >> 4;        // lane-half
  const int l16  = lane & 15;
  const int n0 = blockIdx.y * 16;
  const int m0 = blockIdx.x * 128 + wave * 16;

  __shared__ bf16  ldsB[16 * MAXK];
  __shared__ float s_ps[8][32];
  __shared__ float s_pq[8][32];

  if (wave == 0) {
    // ---- TDM: load Wt rows [n0, n0+16) x K (bf16) row-major into LDS ----
    unsigned long long gaddr = (unsigned long long)(const void*)(Wt + (size_t)n0 * K);
    unsigned laddr = (unsigned)(unsigned long long)(const void*)&ldsB[0]; // low 32b of flat = LDS offset
    u32x4 g0 = { 1u,                                            // count=1 (valid user D#)
                 laddr,                                          // lds_addr
                 (unsigned)(gaddr & 0xFFFFFFFFu),                // global_addr[31:0]
                 ((unsigned)((gaddr >> 32) & 0x01FFFFFFu)) | (2u << 30) }; // addr[56:32] | type=2
    i32x8 g1 = { (int)(1u << 16),                                // data_size=1 (2 bytes)
                 (int)((unsigned)(K & 0xFFFF) << 16),            // tensor_dim0[15:0]
                 (int)(((unsigned)(K >> 16) & 0xFFFFu) | ((unsigned)(C & 0xFFFF) << 16)),
                 (int)(((unsigned)(C >> 16) & 0xFFFFu) | ((unsigned)(K & 0xFFFF) << 16)), // tile_dim0=K
                 16,                                             // tile_dim1=16, tile_dim2=0
                 K,                                              // tensor_dim0_stride lo
                 0, 0 };
    i32x4 g2 = { 0, 0, 0, 0 };
    i32x4 g3 = { 0, 0, 0, 0 };
    i32x8 g4 = { 0, 0, 0, 0, 0, 0, 0, 0 };                       // unused trailing group
    __builtin_amdgcn_tensor_load_to_lds(g0, g1, g2, g3, g4, 0);
    __builtin_amdgcn_s_wait_tensorcnt(0);
  }
  __syncthreads();

  v8f acc = { 0.f, 0.f, 0.f, 0.f, 0.f, 0.f, 0.f, 0.f };
  const bf16* xrow = X + (size_t)(m0 + l16) * K;     // A: row m = m0 + lane%16
  const bf16* brow = &ldsB[l16 * K];                 // B: col n = n0 + lane%16

  for (int kb = 0; kb < K; kb += 32) {
    __builtin_prefetch(xrow + kb + 64, 0, 1);        // global_prefetch_b8
    // A fragment: lane-half 0 -> K {kb+0..7, kb+16..23}; half 1 -> {kb+8..15, kb+24..31}
    bf16x8 a0 = *(const bf16x8*)(xrow + kb + 8 * hi);
    bf16x8 a1 = *(const bf16x8*)(xrow + kb + 16 + 8 * hi);
    v16bf a = __builtin_shufflevector(a0, a1, 0, 1, 2, 3, 4, 5, 6, 7,
                                      8, 9, 10, 11, 12, 13, 14, 15);
    // B fragment: lane-half 0 -> K {kb+0..7, kb+16..23}; half 1 -> {kb+8..15, kb+24..31}
    bf16x8 b0 = *(const bf16x8*)(brow + kb + 8 * hi);
    bf16x8 b1 = *(const bf16x8*)(brow + kb + 16 + 8 * hi);
    v16bf b = __builtin_shufflevector(b0, b1, 0, 1, 2, 3, 4, 5, 6, 7,
                                      8, 9, 10, 11, 12, 13, 14, 15);
    acc = __builtin_amdgcn_wmma_f32_16x16x32_bf16(false, a, false, b,
                                                  (short)0, acc, false, false);
  }

  // Epilogue: bias, store (C/D layout: m = m0 + r + 8*hi, n = n0 + lane%16),
  // and per-column BN partials (sum over this tile's 16 rows, then 8 waves).
  const int n = n0 + l16;
  const float bb = bias ? bias[n] : 0.0f;
  float bsum = 0.f, bsq = 0.f;
#pragma unroll
  for (int r = 0; r < 8; ++r) {
    float v = acc[r] + bb;
    bsum += v; bsq += v * v;
    size_t oi = (size_t)(m0 + r + 8 * hi) * C + n;
    if (Yf32) Yf32[oi] = v;
    else      Y[oi] = (bf16)v;
  }
  s_ps[wave][lane] = bsum;
  s_pq[wave][lane] = bsq;
  __syncthreads();
  if (psum && threadIdx.x < 16) {
    float ts = 0.f, tq = 0.f;
#pragma unroll
    for (int w = 0; w < 8; ++w) {
      ts += s_ps[w][threadIdx.x] + s_ps[w][threadIdx.x + 16];
      tq += s_pq[w][threadIdx.x] + s_pq[w][threadIdx.x + 16];
    }
    psum[(size_t)(n0 + threadIdx.x) * nblkM + blockIdx.x] = ts;
    psq [(size_t)(n0 + threadIdx.x) * nblkM + blockIdx.x] = tq;
  }
}

// -------------------------------- BN finalize + apply ------------------------
__global__ void k_stats_finalize(const float* __restrict__ psum, const float* __restrict__ psq,
                                 int nblk, int C, float invM,
                                 float* __restrict__ mean, float* __restrict__ istd) {
  int c = blockIdx.x * 64 + threadIdx.x;
  if (c >= C) return;
  float s = 0.f, q = 0.f;
  for (int i = 0; i < nblk; ++i) { s += psum[(size_t)c * nblk + i]; q += psq[(size_t)c * nblk + i]; }
  float m = s * invM;
  float v = fmaxf(q * invM - m * m, 0.0f);
  mean[c] = m;
  istd[c] = rsqrtf(v + 1e-5f);
}

__global__ void k_bn_relu(bf16* __restrict__ Y, const float* __restrict__ gamma,
                          const float* __restrict__ beta, const float* __restrict__ mean,
                          const float* __restrict__ istd, int C, long long total) {
  long long t = (long long)blockIdx.x * 256 + threadIdx.x;
  if (t >= total) return;
  int c = (int)(t % C);
  float x = (float)Y[t];
  float y = gamma[c] * (x - mean[c]) * istd[c] + beta[c];
  Y[t] = (bf16)fmaxf(y, 0.0f);
}

// -------------------------------- max over the sample axis -------------------
__global__ void k_maxpool(const bf16* __restrict__ in, float* __restrict__ out,
                          int G, int ns, int C) {
  long long t = (long long)blockIdx.x * 256 + threadIdx.x;
  if (t >= (long long)G * C) return;
  int g = (int)(t / C), c = (int)(t % C);
  float m = -3.4e38f;
  for (int j = 0; j < ns; ++j)
    m = fmaxf(m, (float)in[((size_t)g * ns + j) * C + c]);
  out[t] = m;
}

__global__ void k_outcopy(const float* __restrict__ Y, float* __restrict__ out) {
  int t = blockIdx.x * 256 + threadIdx.x;
  if (t >= BB * 59) return;
  int b = t / 59, c = t % 59;
  out[t] = Y[b * 64 + c];
}

// ---------------------------------------------------------------------------
extern "C" void kernel_launch(void* const* d_in, const int* in_sizes, int n_in,
                              void* d_out, int out_size, void* d_ws, size_t ws_size,
                              hipStream_t stream) {
  (void)in_sizes; (void)n_in; (void)out_size; (void)ws_size;
  const float* pts = (const float*)d_in[0];
  const float* onehot = (const float*)d_in[1];
  int pi = 2;
  const float *saW[9], *saG[9], *saBt[9];
  for (int l = 0; l < 9; ++l) {
    saW[l] = (const float*)d_in[pi++];
    saG[l] = (const float*)d_in[pi++];
    saBt[l] = (const float*)d_in[pi++];
  }
  const float* fc1W = (const float*)d_in[pi++]; const float* fc1b = (const float*)d_in[pi++];
  const float* fc1g = (const float*)d_in[pi++]; const float* fc1be = (const float*)d_in[pi++];
  const float* fc2W = (const float*)d_in[pi++]; const float* fc2b = (const float*)d_in[pi++];
  const float* fc2g = (const float*)d_in[pi++]; const float* fc2be = (const float*)d_in[pi++];
  const float* fc3W = (const float*)d_in[pi++]; const float* fc3b = (const float*)d_in[pi++];

  // ---- workspace carve-up (~830 MB peak) ----
  char* base = (char*)d_ws; size_t off = 0;
  auto alloc = [&](size_t bytes) -> void* {
    void* r = base + off; off = (off + bytes + 255) & ~(size_t)255; return r;
  };
  float* xyz  = (float*)alloc((size_t)BB * NPTS * 3 * 4);
  float* l1x  = (float*)alloc((size_t)BB * 128 * 3 * 4);
  float* l2x  = (float*)alloc((size_t)BB * 32 * 3 * 4);
  int*   fid1 = (int*)alloc((size_t)BB * 128 * 4);
  int*   fid2 = (int*)alloc((size_t)BB * 32 * 4);
  int*   bq1  = (int*)alloc((size_t)BB * 128 * 64 * 4);
  int*   bq2  = (int*)alloc((size_t)BB * 32 * 64 * 4);
  float* l1f  = (float*)alloc((size_t)BB * 128 * 128 * 4);
  float* l2f  = (float*)alloc((size_t)BB * 32 * 256 * 4);
  float* gf   = (float*)alloc((size_t)BB * 512 * 4);
  float* yfc3 = (float*)alloc((size_t)BB * 64 * 4);
  float* meanb = (float*)alloc(512 * 4);
  float* istdb = (float*)alloc(512 * 4);
  float* psum = (float*)alloc((size_t)128 * 16384 * 4);  // max C*nblk over all layers
  float* psq  = (float*)alloc((size_t)128 * 16384 * 4);

  int sa_Cin[9] = { 3, 64, 64, 131, 128, 128, 259, 256, 256 };
  int sa_K[9]   = { 32, 64, 64, 160, 128, 128, 288, 256, 256 };
  int sa_C[9]   = { 64, 64, 128, 128, 128, 256, 256, 256, 512 };
  bf16* Wp[12];
  for (int l = 0; l < 9; ++l) Wp[l] = (bf16*)alloc((size_t)sa_C[l] * sa_K[l] * 2);
  Wp[9]  = (bf16*)alloc((size_t)512 * 544 * 2);
  Wp[10] = (bf16*)alloc((size_t)256 * 512 * 2);
  Wp[11] = (bf16*)alloc((size_t)64 * 256 * 2);
  float* bp1 = (float*)alloc(512 * 4);
  float* bp2 = (float*)alloc(256 * 4);
  float* bp3 = (float*)alloc(64 * 4);
  bf16* bufA = (bf16*)alloc((size_t)2097152 * 64 * 2);   // 256 MB
  bf16* bufB = (bf16*)alloc((size_t)2097152 * 128 * 2);  // 512 MB

  auto cdiv = [](long long a, long long b) { return (int)((a + b - 1) / b); };

  // ---- pack weights & biases into padded bf16 / f32 ----
  for (int l = 0; l < 9; ++l)
    k_wpack<<<cdiv((long long)sa_C[l] * sa_K[l], 256), 256, 0, stream>>>(
        saW[l], Wp[l], sa_C[l], sa_Cin[l], sa_C[l], sa_K[l]);
  k_wpack<<<cdiv(512LL * 544, 256), 256, 0, stream>>>(fc1W, Wp[9], 512, 514, 512, 544);
  k_wpack<<<cdiv(256LL * 512, 256), 256, 0, stream>>>(fc2W, Wp[10], 256, 512, 256, 512);
  k_wpack<<<cdiv(64LL * 256, 256), 256, 0, stream>>>(fc3W, Wp[11], 59, 256, 64, 256);
  k_bpack<<<cdiv(512, 256), 256, 0, stream>>>(fc1b, bp1, 512, 512);
  k_bpack<<<cdiv(256, 256), 256, 0, stream>>>(fc2b, bp2, 256, 256);
  k_bpack<<<cdiv(64, 256), 256, 0, stream>>>(fc3b, bp3, 59, 64);

  auto run_layer = [&](const bf16* Xb, bf16* Yb, float* Yf, const bf16* W_,
                       const float* bias_, const float* g_, const float* be_,
                       int M, int K, int C, bool bn) {
    int nblk = M / 128;
    dim3 grid(nblk, C / 16);
    k_gemm<<<grid, 256, 0, stream>>>(Xb, W_, bias_, Yb, Yf,
                                     bn ? psum : (float*)nullptr, psq, nblk, M, K, C);
    if (bn) {
      k_stats_finalize<<<cdiv(C, 64), 64, 0, stream>>>(psum, psq, nblk, C,
                                                       1.0f / (float)M, meanb, istdb);
      long long tot = (long long)M * C;
      k_bn_relu<<<cdiv(tot, 256), 256, 0, stream>>>(Yb, g_, be_, meanb, istdb, C, tot);
    }
  };

  // ---- SA1 ----
  k_transpose<<<cdiv((long long)BB * NPTS * 3, 256), 256, 0, stream>>>(pts, xyz);
  k_fps<<<BB, 256, 0, stream>>>(xyz, NPTS, 128, fid1);
  k_gather3<<<cdiv((long long)BB * 128 * 3, 256), 256, 0, stream>>>(xyz, fid1, NPTS, 128, l1x);
  k_ballquery<<<cdiv((long long)BB * 128, 256), 256, 0, stream>>>(xyz, l1x, NPTS, 128, 0.04f, 64, bq1);
  k_group_sa1<<<cdiv((long long)BB * 128 * 64, 256), 256, 0, stream>>>(xyz, l1x, bq1, bufA);
  {
    int M = BB * 128 * 64;
    run_layer(bufA, bufB, nullptr, Wp[0], nullptr, saG[0], saBt[0], M, 32, 64, true);
    run_layer(bufB, bufA, nullptr, Wp[1], nullptr, saG[1], saBt[1], M, 64, 64, true);
    run_layer(bufA, bufB, nullptr, Wp[2], nullptr, saG[2], saBt[2], M, 64, 128, true);
    k_maxpool<<<cdiv((long long)BB * 128 * 128, 256), 256, 0, stream>>>(bufB, l1f, BB * 128, 64, 128);
  }

  // ---- SA2 ----
  k_fps<<<BB, 256, 0, stream>>>(l1x, 128, 32, fid2);
  k_gather3<<<cdiv((long long)BB * 32 * 3, 256), 256, 0, stream>>>(l1x, fid2, 128, 32, l2x);
  k_ballquery<<<cdiv((long long)BB * 32, 256), 256, 0, stream>>>(l1x, l2x, 128, 32, 0.16f, 64, bq2);
  k_group_sa2<<<cdiv((long long)BB * 32 * 64, 256), 256, 0, stream>>>(l1x, l2x, bq2, l1f, bufA);
  {
    int M = BB * 32 * 64;
    run_layer(bufA, bufB, nullptr, Wp[3], nullptr, saG[3], saBt[3], M, 160, 128, true);
    run_layer(bufB, bufA, nullptr, Wp[4], nullptr, saG[4], saBt[4], M, 128, 128, true);
    run_layer(bufA, bufB, nullptr, Wp[5], nullptr, saG[5], saBt[5], M, 128, 256, true);
    k_maxpool<<<cdiv((long long)BB * 32 * 256, 256), 256, 0, stream>>>(bufB, l2f, BB * 32, 64, 256);
  }

  // ---- SA3 (group-all) ----
  k_group_sa3<<<cdiv((long long)BB * 32, 256), 256, 0, stream>>>(l2x, l2f, bufA);
  {
    int M = BB * 32;
    run_layer(bufA, bufB, nullptr, Wp[6], nullptr, saG[6], saBt[6], M, 288, 256, true);
    run_layer(bufB, bufA, nullptr, Wp[7], nullptr, saG[7], saBt[7], M, 256, 256, true);
    run_layer(bufA, bufB, nullptr, Wp[8], nullptr, saG[8], saBt[8], M, 256, 512, true);
    k_maxpool<<<cdiv((long long)BB * 512, 256), 256, 0, stream>>>(bufB, gf, BB, 32, 512);
  }

  // ---- FC head ----
  k_fcin<<<cdiv(BB, 64), 64, 0, stream>>>(gf, onehot, bufA);
  run_layer(bufA, bufB, nullptr, Wp[9], bp1, fc1g, fc1be, BB, 544, 512, true);
  run_layer(bufB, bufA, nullptr, Wp[10], bp2, fc2g, fc2be, BB, 512, 256, true);
  run_layer(bufA, bufB, yfc3, Wp[11], bp3, nullptr, nullptr, BB, 256, 64, false);
  k_outcopy<<<cdiv((long long)BB * 59, 256), 256, 0, stream>>>(yfc3, (float*)d_out);
}
